// LlamaRealAttention_59906203844907
// MI455X (gfx1250) — matrix operand order repaired and checked by reading, not compiled, not run on hardware
//
#include <hip/hip_runtime.h>
#include <hip/hip_bf16.h>

// ---------------------------------------------------------------------------
// LlamaRealAttention for MI455X (gfx1250), wave32 + v_wmma_f32_16x16x32_f16.
// Output is ~1.06 GB (attn_weights + reweight_mask + attn_output), so the
// pipeline is HBM-write bound (~23.3 TB/s floor). f16 WMMA w/ fp32 accumulate
// keeps the ~77 GFLOP of matmul work far below the memory floor.
// GEMM weights (B panel, shared by all 8 waves of a block) are staged through
// LDS by the Tensor Data Mover (tensor_load_to_lds, TENSORcnt, double buffer)
// to kill redundant L2 traffic and overlap DMA with WMMA.
// ---------------------------------------------------------------------------

#define S_LEN  2048
#define DMODEL 2048
#define NH     32
#define NKV    8
#define HD     64
#define SCALE_QK 0.125f     // 64^-0.5
#define NEGV   (-1.0e9f)

typedef _Float16 h16 __attribute__((ext_vector_type(16)));
typedef _Float16 h8  __attribute__((ext_vector_type(8)));
typedef float    f8  __attribute__((ext_vector_type(8)));
typedef float    f4  __attribute__((ext_vector_type(4)));
typedef unsigned int u32x4 __attribute__((ext_vector_type(4)));
typedef unsigned int u32x8 __attribute__((ext_vector_type(8)));

// ---- WMMA fragment helpers (layouts per CDNA5 ISA 7.12.2, wave32) ----------
// A (16x32 f16, M x K): lanes 0-15 = rows, hold K{0..7,16..23}; lanes 16-31
// hold K{8..15,24..31}.
__device__ __forceinline__ h16 load_a_frag(const _Float16* A, int lda, int row0,
                                           int k, int lane) {
  const int row = row0 + (lane & 15);
  const int sel = (lane >> 4) & 1;
  const _Float16* p = A + (size_t)row * lda + k + sel * 8;
  h8 lo = *(const h8*)(p);
  h8 hi = *(const h8*)(p + 16);
  h16 r;
#pragma unroll
  for (int i = 0; i < 8; ++i) { r[i] = lo[i]; r[i + 8] = hi[i]; }
  return r;
}

// B (32x16 f16, K x N): lane n%16 holds column n; lanes 0-15 hold
// K{0..7,16..23}, lanes 16-31 hold K{8..15,24..31} (same as A).
__device__ __forceinline__ h16 load_b_frag(const _Float16* B, int ldb, int col0,
                                           int k, int lane) {
  const int col = col0 + (lane & 15);
  const int kb  = k + ((lane >> 4) & 1) * 8;
  const _Float16* p = B + (size_t)col * ldb + kb;
  h8 lo = *(const h8*)(p);
  h8 hi = *(const h8*)(p + 16);
  h16 r;
#pragma unroll
  for (int i = 0; i < 8; ++i) { r[i] = lo[i]; r[i + 8] = hi[i]; }
  return r;
}

__device__ __forceinline__ f8 wmma16(h16 a, h16 b, f8 c) {
  return __builtin_amdgcn_wmma_f32_16x16x32_f16(false, a, false, b, (short)0, c,
                                                false, false);
}

__device__ __forceinline__ float wave_sum(float v) {
#pragma unroll
  for (int m = 16; m > 0; m >>= 1) v += __shfl_xor(v, m, 32);
  return v;
}
__device__ __forceinline__ float wave_max(float v) {
#pragma unroll
  for (int m = 16; m > 0; m >>= 1) v = fmaxf(v, __shfl_xor(v, m, 32));
  return v;
}

__device__ __forceinline__ void wait_tensorcnt0() {
#if defined(__has_builtin)
#if __has_builtin(__builtin_amdgcn_s_wait_tensorcnt)
  __builtin_amdgcn_s_wait_tensorcnt(0);
#else
  asm volatile("s_wait_tensorcnt 0x0" ::: "memory");
#endif
#else
  asm volatile("s_wait_tensorcnt 0x0" ::: "memory");
#endif
}

// ---- TDM: DMA one 64-row x 128-half B tile (row stride K halves) into LDS --
// D# per CDNA5 ISA ch.8: group0 = {count/type/lds_addr/global_addr},
// group1 = {data_size=2B, tensor dims, tile 128x64, dim0 stride}.
__device__ __forceinline__ void tdm_load_b_tile(const _Float16* gsrc,
                                                unsigned lds_off, unsigned K) {
  const unsigned long long ga = (unsigned long long)(size_t)gsrc;
  u32x4 g0;
  g0[0] = 1u;                                   // count=1 (user descriptor)
  g0[1] = lds_off;                              // D#.lds_addr (bytes)
  g0[2] = (unsigned)(ga & 0xFFFFFFFFu);         // global_addr[31:0]
  g0[3] = (unsigned)((ga >> 32) & 0x01FFFFFFu)  // global_addr[56:32]
          | (2u << 30);                         // type=2 ("image")
  u32x8 g1;
  g1[0] = (1u << 16);                           // data_size=1 -> 2 bytes
  g1[1] = (K & 0xFFFFu) << 16;                  // tensor_dim0[15:0]
  g1[2] = (K >> 16) | (64u << 16);              // tensor_dim0 hi | tensor_dim1
  g1[3] = (128u << 16);                         // tile_dim0 = 128 elements
  g1[4] = 64u;                                  // tile_dim1 = 64 rows
  g1[5] = K;                                    // tensor_dim0_stride (elems)
  g1[6] = 0u;
  g1[7] = 0u;
  asm volatile("tensor_load_to_lds %0, %1" :: "s"(g0), "s"(g1) : "memory");
}

// ---- fp32 -> f16 convert ---------------------------------------------------
__global__ void __launch_bounds__(256) cvt_kernel(const float* __restrict__ src,
                                                  _Float16* __restrict__ dst,
                                                  int n) {
  int i = blockIdx.x * 256 + threadIdx.x;
  if (i < n) dst[i] = (_Float16)src[i];
}

// ---- NT GEMM: C(MxN,f32) = A(MxK,f16) * B(NxK,f16)^T -----------------------
// Block = 8 waves, 128x64 block tile; each wave owns a 32x32 output tile.
// B panel staged through LDS by the TDM, double-buffered per 128-k chunk.
__global__ void __launch_bounds__(256)
gemm_nt_kernel(const _Float16* __restrict__ A, const _Float16* __restrict__ B,
               float* __restrict__ C, int M, int N, int K) {
  __shared__ _Float16 Bs[2][64 * 128];          // 2 x 16 KiB
  const int lane = threadIdx.x & 31;
  const int wid  = threadIdx.x >> 5;
  const int m0 = blockIdx.y * 128 + (wid >> 1) * 32;
  const int nblk = blockIdx.x * 64;             // block's B panel origin
  const int n0 = (wid & 1) * 32;                // n offset inside LDS tile
  f8 acc00 = {}, acc01 = {}, acc10 = {}, acc11 = {};

  const int nChunks = K >> 7;                   // 128-k chunks
  if (wid == 0)
    tdm_load_b_tile(B + (size_t)nblk * K, (unsigned)(size_t)&Bs[0][0], K);

  for (int c = 0; c < nChunks; ++c) {
    if (wid == 0) wait_tensorcnt0();            // chunk c landed in LDS
    __syncthreads();
    if (wid == 0 && c + 1 < nChunks)            // overlap DMA with compute
      tdm_load_b_tile(B + (size_t)nblk * K + (c + 1) * 128,
                      (unsigned)(size_t)&Bs[(c + 1) & 1][0], K);
    const _Float16* Bl = &Bs[c & 1][0];
#pragma unroll
    for (int kk = 0; kk < 4; ++kk) {
      const int k = (c << 7) + kk * 32;
      h16 a0 = load_a_frag(A, K, m0,      k, lane);
      h16 a1 = load_a_frag(A, K, m0 + 16, k, lane);
      h16 b0 = load_b_frag(Bl, 128, n0,      kk * 32, lane);
      h16 b1 = load_b_frag(Bl, 128, n0 + 16, kk * 32, lane);
      if (k + 128 < K)  // speculative L2 prefetch of upcoming A rows
        __builtin_prefetch(A + (size_t)(m0 + (lane & 15)) * K + k + 128, 0, 1);
      acc00 = wmma16(a0, b0, acc00);
      acc01 = wmma16(a0, b1, acc01);
      acc10 = wmma16(a1, b0, acc10);
      acc11 = wmma16(a1, b1, acc11);
    }
    __syncthreads();                            // buffer reuse fence
  }

  const int mo = (lane >> 4) * 8;
  const int nn = lane & 15;
  const int nc = nblk + n0;
#pragma unroll
  for (int r = 0; r < 8; ++r) {
    C[(size_t)(m0 + mo + r) * N + nc + nn]           = acc00[r];
    C[(size_t)(m0 + mo + r) * N + nc + 16 + nn]      = acc01[r];
    C[(size_t)(m0 + 16 + mo + r) * N + nc + nn]      = acc10[r];
    C[(size_t)(m0 + 16 + mo + r) * N + nc + 16 + nn] = acc11[r];
  }
}

// ---- RoPE on Q + qs = <q_rope, rw_q> (one wave per (s,h)) ------------------
__global__ void __launch_bounds__(256)
rope_q_kernel(const float* __restrict__ Qf, const float* __restrict__ cosb,
              const float* __restrict__ sinb, const float* __restrict__ rwq,
              _Float16* __restrict__ Qh, float* __restrict__ qs) {
  const int lane = threadIdx.x & 31;
  const int wid  = threadIdx.x >> 5;
  const int wl = blockIdx.x * 8 + wid;     // 0 .. S*H-1
  const int s = wl >> 5;
  const int h = wl & 31;
  const float* q = Qf + (size_t)s * DMODEL + h * HD;
  const float lo = q[lane], hi = q[lane + 32];
  const float c0 = cosb[s * HD + lane], c1 = cosb[s * HD + lane + 32];
  const float s0 = sinb[s * HD + lane], s1 = sinb[s * HD + lane + 32];
  const float rlo = lo * c0 - hi * s0;   // rotate_half
  const float rhi = hi * c1 + lo * s1;
  _Float16* qo = Qh + (size_t)s * DMODEL + h * HD;
  qo[lane] = (_Float16)rlo;
  qo[lane + 32] = (_Float16)rhi;
  float part = rlo * rwq[h * HD + lane] + rhi * rwq[h * HD + lane + 32];
  part = wave_sum(part);
  if (lane == 0) qs[h * S_LEN + s] = part;
}

// ---- RoPE on K + ks[h] = <k_rope(kv=h/4), rw_k[h]> (wave per (s,kv)) -------
__global__ void __launch_bounds__(256)
rope_k_kernel(const float* __restrict__ Kf, const float* __restrict__ cosb,
              const float* __restrict__ sinb, const float* __restrict__ rwk,
              _Float16* __restrict__ Kh, float* __restrict__ ks) {
  const int lane = threadIdx.x & 31;
  const int wid  = threadIdx.x >> 5;
  const int wl = blockIdx.x * 8 + wid;     // 0 .. S*NKV-1
  const int s  = wl >> 3;
  const int kv = wl & 7;
  const float* kp = Kf + (size_t)s * (NKV * HD) + kv * HD;
  const float lo = kp[lane], hi = kp[lane + 32];
  const float c0 = cosb[s * HD + lane], c1 = cosb[s * HD + lane + 32];
  const float s0 = sinb[s * HD + lane], s1 = sinb[s * HD + lane + 32];
  const float rlo = lo * c0 - hi * s0;
  const float rhi = hi * c1 + lo * s1;
  _Float16* ko = Kh + (size_t)s * (NKV * HD) + kv * HD;
  ko[lane] = (_Float16)rlo;
  ko[lane + 32] = (_Float16)rhi;
#pragma unroll
  for (int g = 0; g < 4; ++g) {           // 4 q-heads share this kv head
    const int h = kv * 4 + g;
    float part = rlo * rwk[h * HD + lane] + rhi * rwk[h * HD + lane + 32];
    part = wave_sum(part);
    if (lane == 0) ks[h * S_LEN + s] = part;
  }
}

// ---- V: (s, kv*64+d) f32 -> Vt (kv*64+d, s) f16 (K-contiguous for WMMA B) --
__global__ void __launch_bounds__(256)
vtrans_kernel(const float* __restrict__ Vf, _Float16* __restrict__ Vt) {
  const int idx = blockIdx.x * 256 + threadIdx.x;  // c*S + s
  const int c = idx >> 11;
  const int s = idx & (S_LEN - 1);
  Vt[idx] = (_Float16)Vf[(size_t)s * (NKV * HD) + c];
}

// ---- tok[s] = rw_tok[input_ids[s]] -----------------------------------------
__global__ void __launch_bounds__(256)
tok_kernel(const int* __restrict__ ids, const float* __restrict__ rwtok,
           float* __restrict__ tok) {
  int i = blockIdx.x * 256 + threadIdx.x;
  if (i < S_LEN) {
    int v = ids[i];
    if (v < 0 || v >= 32000) v = 0;
    tok[i] = rwtok[v];
  }
}

// ---- scores: q@k^T * scale + tanh(qs+ks+tok)*mask, causal; writes W and RW -
__global__ void __launch_bounds__(256)
scores_kernel(const _Float16* __restrict__ Qh, const _Float16* __restrict__ Kh,
              const float* __restrict__ qs, const float* __restrict__ ks,
              const float* __restrict__ tok, const float* __restrict__ mask,
              float* __restrict__ W, float* __restrict__ RW) {
  const int lane = threadIdx.x & 31;
  const int wid  = threadIdx.x >> 5;
  const int h  = blockIdx.z;
  const int q0 = blockIdx.y * 16;
  const int k0 = blockIdx.x * 512 + wid * 64;
  const _Float16* Qb = Qh + h * HD;          // row stride DMODEL, cols = d
  const _Float16* Kb = Kh + (h >> 2) * HD;   // GQA: kv head = h/4
  h16 a0 = load_a_frag(Qb, DMODEL, q0, 0, lane);
  h16 a1 = load_a_frag(Qb, DMODEL, q0, 32, lane);
  const int mo = (lane >> 4) * 8;
  const int nn = lane & 15;
  float qsv[8];
#pragma unroll
  for (int r = 0; r < 8; ++r) qsv[r] = qs[h * S_LEN + q0 + mo + r];
#pragma unroll
  for (int t = 0; t < 4; ++t) {
    const int kt = k0 + t * 16;
    h16 b0 = load_b_frag(Kb, NKV * HD, kt, 0, lane);
    h16 b1 = load_b_frag(Kb, NKV * HD, kt, 32, lane);
    f8 acc = {};
    acc = wmma16(a0, b0, acc);
    acc = wmma16(a1, b1, acc);
    const int kk = kt + nn;
    const float ksv = ks[h * S_LEN + kk];
    const float tv  = tok[kk];
    const float mv  = mask[kk];
#pragma unroll
    for (int r = 0; r < 8; ++r) {
      const int qr = q0 + mo + r;
      const float rw = tanhf(qsv[r] + ksv + tv) * mv;
      const size_t idx = ((size_t)h * S_LEN + qr) * S_LEN + kk;
      RW[idx] = rw;                              // full, no causal mask
      W[idx]  = (kk <= qr) ? (acc[r] * SCALE_QK + rw) : NEGV;
    }
  }
}

// ---- softmax in place over each (h,q) row of W (wave per row) --------------
__global__ void __launch_bounds__(256)
softmax_kernel(float* __restrict__ W) {
  const int lane = threadIdx.x & 31;
  const int wid  = threadIdx.x >> 5;
  float* row = W + ((size_t)blockIdx.x * 8 + wid) * S_LEN;
  f4 v[16];
  float mx = -3.0e38f;
#pragma unroll
  for (int i = 0; i < 16; ++i) {
    v[i] = *(const f4*)(row + i * 128 + lane * 4);
    mx = fmaxf(mx, fmaxf(fmaxf(v[i][0], v[i][1]), fmaxf(v[i][2], v[i][3])));
  }
  mx = wave_max(mx);
  float sum = 0.f;
#pragma unroll
  for (int i = 0; i < 16; ++i)
#pragma unroll
    for (int j = 0; j < 4; ++j) { v[i][j] = __expf(v[i][j] - mx); sum += v[i][j]; }
  sum = wave_sum(sum);
  const float inv = 1.0f / sum;
#pragma unroll
  for (int i = 0; i < 16; ++i) {
    v[i] *= inv;
    *(f4*)(row + i * 128 + lane * 4) = v[i];
  }
}

// ---- out = attn_weights @ V (wave: one q-tile x full D=64, causal-bounded) -
__global__ void __launch_bounds__(256)
attnv_kernel(const float* __restrict__ W, const _Float16* __restrict__ Vt,
             _Float16* __restrict__ Oh) {
  const int lane = threadIdx.x & 31;
  const int wid  = threadIdx.x >> 5;
  const int wl = blockIdx.x * 8 + wid;      // 0 .. NH*(S/16)-1
  const int h  = wl >> 7;
  const int q0 = (wl & 127) * 16;
  const float* P = W + (size_t)h * S_LEN * S_LEN;
  const _Float16* Vb = Vt + (size_t)(h >> 2) * HD * S_LEN;
  f8 acc0 = {}, acc1 = {}, acc2 = {}, acc3 = {};
  const int row = q0 + (lane & 15);
  const int sel = (lane >> 4) & 1;
  for (int k = 0; k < q0 + 16; k += 32) {   // P[k>q] == 0 after softmax
    const float* pr = P + (size_t)row * S_LEN + k + sel * 8;
    f4 f0 = *(const f4*)(pr);
    f4 f1 = *(const f4*)(pr + 4);
    f4 f2 = *(const f4*)(pr + 16);
    f4 f3 = *(const f4*)(pr + 20);
    h16 a;
#pragma unroll
    for (int i = 0; i < 4; ++i) {
      a[i]      = (_Float16)f0[i];
      a[4 + i]  = (_Float16)f1[i];
      a[8 + i]  = (_Float16)f2[i];
      a[12 + i] = (_Float16)f3[i];
    }
    h16 b0 = load_b_frag(Vb, S_LEN, 0,  k, lane);
    h16 b1 = load_b_frag(Vb, S_LEN, 16, k, lane);
    h16 b2 = load_b_frag(Vb, S_LEN, 32, k, lane);
    h16 b3 = load_b_frag(Vb, S_LEN, 48, k, lane);
    acc0 = wmma16(a, b0, acc0);
    acc1 = wmma16(a, b1, acc1);
    acc2 = wmma16(a, b2, acc2);
    acc3 = wmma16(a, b3, acc3);
  }
  const int mo = (lane >> 4) * 8;
  const int nn = lane & 15;
#pragma unroll
  for (int r = 0; r < 8; ++r) {
    _Float16* o = Oh + (size_t)(q0 + mo + r) * DMODEL + h * HD;
    o[nn]      = (_Float16)acc0[r];
    o[16 + nn] = (_Float16)acc1[r];
    o[32 + nn] = (_Float16)acc2[r];
    o[48 + nn] = (_Float16)acc3[r];
  }
}

// ---------------------------------------------------------------------------
extern "C" void kernel_launch(void* const* d_in, const int* in_sizes, int n_in,
                              void* d_out, int out_size, void* d_ws,
                              size_t ws_size, hipStream_t stream) {
  const int*   ids   = (const int*)d_in[0];
  const float* hid   = (const float*)d_in[1];
  const float* cosb  = (const float*)d_in[2];
  const float* sinb  = (const float*)d_in[3];
  const float* mask  = (const float*)d_in[4];
  const float* Wq    = (const float*)d_in[5];
  const float* Wk    = (const float*)d_in[6];
  const float* Wv    = (const float*)d_in[7];
  const float* Wo    = (const float*)d_in[8];
  const float* rwq   = (const float*)d_in[9];
  const float* rwk   = (const float*)d_in[10];
  const float* rwtok = (const float*)d_in[11];

  // d_out: attn_output (S*DM) | attn_weights (H*S*S) | reweight_mask (H*S*S)
  float* attn_out = (float*)d_out;
  float* Wp  = attn_out + (size_t)S_LEN * DMODEL;
  float* RWp = Wp + (size_t)NH * S_LEN * S_LEN;

  // workspace layout (bytes)
  char* wsb = (char*)d_ws;
  _Float16* hid_h = (_Float16*)(wsb);                                  // 8 MiB
  _Float16* wq_h  = (_Float16*)(wsb + ((size_t)8  << 20));             // 8 MiB
  _Float16* wk_h  = (_Float16*)(wsb + ((size_t)16 << 20));             // 2 MiB
  _Float16* wv_h  = (_Float16*)(wsb + ((size_t)18 << 20));             // 2 MiB
  _Float16* wo_h  = (_Float16*)(wsb + ((size_t)20 << 20));             // 8 MiB
  float*    Qf    = (float*)   (wsb + ((size_t)28 << 20));             // 16 MiB
  float*    Kf    = (float*)   (wsb + ((size_t)44 << 20));             // 4 MiB
  float*    Vf    = (float*)   (wsb + ((size_t)48 << 20));             // 4 MiB
  _Float16* Qh    = (_Float16*)(wsb + ((size_t)52 << 20));             // 8 MiB
  _Float16* Kh    = (_Float16*)(wsb + ((size_t)60 << 20));             // 2 MiB
  _Float16* Vt    = (_Float16*)(wsb + ((size_t)62 << 20));             // 2 MiB
  float*    qsb   = (float*)   (wsb + ((size_t)64 << 20));             // 256 KiB
  float*    ksb   = (float*)   (wsb + ((size_t)64 << 20) + (256u << 10));
  float*    tokb  = (float*)   (wsb + ((size_t)64 << 20) + (512u << 10));
  _Float16* outh  = (_Float16*)(wsb + ((size_t)65 << 20));             // 8 MiB

  const int nHid = S_LEN * DMODEL;          // 4M
  const int nWkv = NKV * HD * DMODEL;       // 1M

  // 1) fp32 -> f16 operand staging
  cvt_kernel<<<nHid / 256, 256, 0, stream>>>(hid, hid_h, nHid);
  cvt_kernel<<<nHid / 256, 256, 0, stream>>>(Wq, wq_h, nHid);
  cvt_kernel<<<nWkv / 256, 256, 0, stream>>>(Wk, wk_h, nWkv);
  cvt_kernel<<<nWkv / 256, 256, 0, stream>>>(Wv, wv_h, nWkv);
  cvt_kernel<<<nHid / 256, 256, 0, stream>>>(Wo, wo_h, nHid);

  // 2) Q/K/V projections (NT WMMA GEMMs, TDM-staged B)
  gemm_nt_kernel<<<dim3(DMODEL / 64, S_LEN / 128), 256, 0, stream>>>(
      hid_h, wq_h, Qf, S_LEN, DMODEL, DMODEL);
  gemm_nt_kernel<<<dim3((NKV * HD) / 64, S_LEN / 128), 256, 0, stream>>>(
      hid_h, wk_h, Kf, S_LEN, NKV * HD, DMODEL);
  gemm_nt_kernel<<<dim3((NKV * HD) / 64, S_LEN / 128), 256, 0, stream>>>(
      hid_h, wv_h, Vf, S_LEN, NKV * HD, DMODEL);

  // 3) RoPE + qs/ks; V transpose; tok gather
  rope_q_kernel<<<(S_LEN * NH) / 8, 256, 0, stream>>>(Qf, cosb, sinb, rwq, Qh, qsb);
  rope_k_kernel<<<(S_LEN * NKV) / 8, 256, 0, stream>>>(Kf, cosb, sinb, rwk, Kh, ksb);
  vtrans_kernel<<<(NKV * HD * S_LEN) / 256, 256, 0, stream>>>(Vf, Vt);
  tok_kernel<<<(S_LEN + 255) / 256, 256, 0, stream>>>(ids, rwtok, tokb);

  // 4) scores + reweight_mask (writes both d_out S x S tensors)
  scores_kernel<<<dim3(S_LEN / 512, S_LEN / 16, NH), 256, 0, stream>>>(
      Qh, Kh, qsb, ksb, tokb, mask, Wp, RWp);

  // 5) softmax in place over attn_weights
  softmax_kernel<<<(NH * S_LEN) / 8, 256, 0, stream>>>(Wp);

  // 6) out = P @ V (f16 staging for the output projection)
  attnv_kernel<<<(NH * (S_LEN / 16)) / 8, 256, 0, stream>>>(Wp, Vt, outh);

  // 7) attn_output = out @ Wo^T
  gemm_nt_kernel<<<dim3(DMODEL / 64, S_LEN / 128), 256, 0, stream>>>(
      outh, wo_h, attn_out, S_LEN, DMODEL, DMODEL);
}
